// GaussianRasterizer_16604343566651
// MI455X (gfx1250) — matrix operand order repaired and checked by reading, not compiled, not run on hardware
//
#include <hip/hip_runtime.h>

typedef __attribute__((ext_vector_type(16))) _Float16 v16h;
typedef __attribute__((ext_vector_type(8)))  float    v8f;
typedef __attribute__((ext_vector_type(4)))  float    v4f;

#define MAXN 2048

struct Q16 { v4f q[4]; };  // 16 consecutive-ish floats: runs [0..7] and [16..23]

__device__ __forceinline__ Q16 ld16(const float* base0) {
    Q16 r;
    r.q[0] = *(const v4f*)(base0);        // g offsets 0..3
    r.q[1] = *(const v4f*)(base0 + 4);    // g offsets 4..7
    r.q[2] = *(const v4f*)(base0 + 16);   // g offsets 16..19
    r.q[3] = *(const v4f*)(base0 + 20);   // g offsets 20..23
    return r;
}

__global__ __launch_bounds__(256) void splat_wmma_kernel(
    const float* __restrict__ opacity,
    const float* __restrict__ means,
    const float* __restrict__ stds,
    const float* __restrict__ rhos,
    const float* __restrict__ colors,
    const int* __restrict__ hptr,
    const int* __restrict__ wptr,
    const int* __restrict__ sptr,
    const int* __restrict__ rptr,
    float* __restrict__ out,
    int N)
{
    // LDS: 5*8KB SoA params + 12KB colors(f16, opacity-premultiplied, channel-major) + 8KB bbox = 60KB
    __shared__ alignas(32) float s_mx[MAXN], s_my[MAXN];
    __shared__ alignas(32) float s_ia[MAXN], s_ib2[MAXN], s_ic[MAXN];
    __shared__ alignas(32) _Float16 s_colT[3 * MAXN];   // [ch][g] = opacity*color
    __shared__ unsigned int s_bb[MAXN];                 // xmin|xmax<<8|ymin<<16|ymax<<24

    const int   W       = wptr[0];
    const int   H       = hptr[0];
    const float scale   = (float)sptr[0];
    const float rr      = (float)rptr[0];
    const float cutoff2 = rr * rr;

    const int tid = threadIdx.x;
    #pragma unroll
    for (int j = 0; j < MAXN / 256; ++j) {
        int g = tid + j * 256;
        if (g < N) {
            float mx  = means[2 * g];
            float my  = means[2 * g + 1];
            float sx  = stds[2 * g] * scale;
            float sy  = stds[2 * g + 1] * scale;
            float rho = rhos[g];
            float inv = 1.0f / (1.0f - rho * rho);
            s_mx[g]  = mx;
            s_my[g]  = my;
            s_ia[g]  = inv / (sx * sx);
            s_ib2[g] = -2.0f * rho * inv / (sx * sy);
            s_ic[g]  = inv / (sy * sy);
            float op = opacity[g];
            s_colT[0 * MAXN + g] = (_Float16)(op * colors[3 * g + 0]);
            s_colT[1 * MAXN + g] = (_Float16)(op * colors[3 * g + 1]);
            s_colT[2 * MAXN + g] = (_Float16)(op * colors[3 * g + 2]);
            float rx = rr * sx, ry = rr * sy;
            int xmin = max(0, min(W - 1, (int)(mx - rx) - 1));
            int xmax = max(0, min(W - 1, (int)(mx + rx) + 1));
            int ymin = max(0, min(H - 1, (int)(my - ry) - 1));
            int ymax = max(0, min(H - 1, (int)(my + ry) + 1));
            s_bb[g] = (unsigned)xmin | ((unsigned)xmax << 8) |
                      ((unsigned)ymin << 16) | ((unsigned)ymax << 24);
        } else {
            s_mx[g] = 0.f; s_my[g] = 0.f;
            s_ia[g] = 0.f; s_ib2[g] = 0.f; s_ic[g] = 0.f;
            s_colT[0 * MAXN + g] = (_Float16)0.0f;
            s_colT[1 * MAXN + g] = (_Float16)0.0f;
            s_colT[2 * MAXN + g] = (_Float16)0.0f;
            s_bb[g] = 0x00FF00FFu;                     // empty bbox (min>max)
        }
    }
    __syncthreads();

    // One wave32 per 16-pixel horizontal strip.
    const int lane  = tid & 31;
    const int wave  = tid >> 5;
    const int spr   = W >> 4;
    const int strip = blockIdx.x * (blockDim.x >> 5) + wave;
    const int y     = strip / spr;
    const int x0    = (strip - y * spr) << 4;

    const int  m      = lane & 15;
    const bool hiHalf = lane >= 16;
    const float px = (float)(x0 + m) + 0.5f;
    const float py = (float)y + 0.5f;

    const int abase = hiHalf ? 8 : 0;     // A K-grouping per ISA layout
    const int bbase = hiHalf ? 8 : 0;     // B K-grouping per ISA layout
    const int n     = m;                  // D column = color channel
    const int cn    = (n < 3) ? n : 0;    // cols >=3 read ch0; their D columns are discarded

    const float NEG_INF = -__builtin_inff();
    v8f acc = {};

    for (int k0 = 0; k0 < MAXN; k0 += 32) {
        // ---- wave-uniform chunk cull: lane L tests gaussian k0+L vs this strip ----
        unsigned bb = s_bb[k0 + lane];
        int xmin = (int)(bb & 255u), xmax = (int)((bb >> 8) & 255u);
        int ymin = (int)((bb >> 16) & 255u), ymax = (int)(bb >> 24);
        bool hit = (y >= ymin) & (y <= ymax) & (x0 <= xmax) & (x0 + 15 >= xmin);
        if (__builtin_amdgcn_ballot_w32(hit) == 0u)
            continue;                                   // uniform branch: EXEC stays full

        // ---- SoA param fetch: 5 params x 4 broadcast ds_load_b128 ----
        const int gb = k0 + abase;
        Q16 qmx = ld16(&s_mx[gb]);
        Q16 qmy = ld16(&s_my[gb]);
        Q16 qia = ld16(&s_ia[gb]);
        Q16 qib = ld16(&s_ib2[gb]);
        Q16 qic = ld16(&s_ic[gb]);

        // ---- A tile: 16 pixels x 32 gaussians, branch-free alpha ----
        v16h a;
        #pragma unroll
        for (int i = 0; i < 16; ++i) {
            float mx  = qmx.q[i >> 2][i & 3];
            float my  = qmy.q[i >> 2][i & 3];
            float ia  = qia.q[i >> 2][i & 3];
            float ib2 = qib.q[i >> 2][i & 3];
            float ic  = qic.q[i >> 2][i & 3];
            float dx = px - mx;
            float dy = py - my;
            float m2 = ia * dx * dx + ib2 * dx * dy + ic * dy * dy;
            float t  = (m2 <= cutoff2) ? (-0.72134752f * m2) : NEG_INF;  // exp2(-inf)=0
            a[i] = (_Float16)__builtin_amdgcn_exp2f(t);
        }
        // ---- B tile: f16 runs [0..7] and [16..23] (opacity-premultiplied colors), 2x ds_load_b128 ----
        typedef __attribute__((ext_vector_type(8))) _Float16 v8h_;
        v8h_ b0 = *(const v8h_*)&s_colT[cn * MAXN + k0 + bbase];
        v8h_ b1 = *(const v8h_*)&s_colT[cn * MAXN + k0 + bbase + 16];
        v16h b = __builtin_shufflevector(b0, b1, 0,1,2,3,4,5,6,7,8,9,10,11,12,13,14,15);

        acc = __builtin_amdgcn_wmma_f32_16x16x32_f16(
            false, a, false, b, (short)0, acc, false, false);
    }

    // D layout: lane holds column n; VGPR v = pixel row (v or 8+v).
    if (n < 3) {
        const int mbase = hiHalf ? 8 : 0;
        #pragma unroll
        for (int v = 0; v < 8; ++v) {
            int pm = mbase + v;
            out[(y * W + x0 + pm) * 3 + n] = acc[v];
        }
    }
}

extern "C" void kernel_launch(void* const* d_in, const int* in_sizes, int n_in,
                              void* d_out, int out_size, void* d_ws, size_t ws_size,
                              hipStream_t stream) {
    const float* opacity = (const float*)d_in[0];
    const float* means   = (const float*)d_in[1];
    const float* stds    = (const float*)d_in[2];
    const float* rhos    = (const float*)d_in[3];
    const float* colors  = (const float*)d_in[4];
    const int* hptr = (const int*)d_in[5];
    const int* wptr = (const int*)d_in[6];
    const int* sptr = (const int*)d_in[7];
    const int* rptr = (const int*)d_in[8];
    float* out = (float*)d_out;

    int N = in_sizes[0];
    int pixels = out_size / 3;   // H*W
    int strips = pixels / 16;    // 16 pixels per wave32
    int blocks = strips / 8;     // 8 waves per 256-thread block

    splat_wmma_kernel<<<blocks, 256, 0, stream>>>(opacity, means, stds, rhos, colors,
                                                  hptr, wptr, sptr, rptr, out, N);
}
